// DIN_Attention_67748814127233
// MI455X (gfx1250) — compile-verified, matrix operand order checked by reading, not run on hardware
//
#include <hip/hip_runtime.h>

typedef _Float16 h4_t  __attribute__((ext_vector_type(4)));
typedef _Float16 h8_t  __attribute__((ext_vector_type(8)));
typedef _Float16 h16_t __attribute__((ext_vector_type(16)));
typedef float    f8_t  __attribute__((ext_vector_type(8)));

#define NROWS 819200      // B*S
#define BATCH 4096
#define SEQ   200
#define EMB   64
#define DIN   256
#define H1    64
#define H2    32
#define ROWS_PER_BLOCK 128

__device__ __forceinline__ void lds_fadd(float* p, float v) {
    __hip_atomic_fetch_add(p, v, __ATOMIC_RELAXED, __HIP_MEMORY_SCOPE_WORKGROUP);
}
__device__ __forceinline__ void glb_fadd(float* p, float v) {
    __hip_atomic_fetch_add(p, v, __ATOMIC_RELAXED, __HIP_MEMORY_SCOPE_AGENT);
}

// -------------------------------------------------------------------------
// GEMM1: h1[N,64] = concat(q, f, q-f, q*f)[N,256] @ W1[256,64] + b1
// h1 stored as f16 (halves HBM traffic); BN stats accumulated in f32
// from the un-truncated accumulators.
// -------------------------------------------------------------------------
__global__ __launch_bounds__(256) void gemm1_kernel(
    const float* __restrict__ query, const float* __restrict__ facts,
    const float* __restrict__ W1, const float* __restrict__ bias1,
    _Float16* __restrict__ h1, float* __restrict__ stats1)
{
    __shared__ _Float16 Qs[ROWS_PER_BLOCK][72];
    __shared__ _Float16 Fs[ROWS_PER_BLOCK][72];   // reused as f16 output tile
    __shared__ _Float16 Wt[H1][264];              // transposed W1: Wt[col][k]
    __shared__ float ssum[H1];
    __shared__ float ssq[H1];

    const int tid = threadIdx.x;
    const int n0  = blockIdx.x * ROWS_PER_BLOCK;

    __builtin_prefetch(facts + (size_t)n0 * EMB, 0, 0);

    if (tid < H1) { ssum[tid] = 0.0f; ssq[tid] = 0.0f; }

    // stage Fs / Qs (f32 -> f16), 128 rows x 64 cols each
    for (int i = tid; i < ROWS_PER_BLOCK * 16; i += 256) {
        int r  = i >> 4;
        int c4 = i & 15;
        int c  = c4 * 4;
        const float4 fv = ((const float4*)(facts + (size_t)(n0 + r) * EMB))[c4];
        int b = (n0 + r) / SEQ;
        const float4 qv = ((const float4*)(query + (size_t)b * EMB))[c4];
        h4_t fh = { (_Float16)fv.x, (_Float16)fv.y, (_Float16)fv.z, (_Float16)fv.w };
        h4_t qh = { (_Float16)qv.x, (_Float16)qv.y, (_Float16)qv.z, (_Float16)qv.w };
        *(h4_t*)(&Fs[r][c]) = fh;
        *(h4_t*)(&Qs[r][c]) = qh;
    }
    // stage W1 transposed as f16
    for (int i = tid; i < DIN * H1; i += 256) {
        int k = i >> 6;
        int c = i & 63;
        Wt[c][k] = (_Float16)W1[i];
    }
    __syncthreads();

    const int wave   = tid >> 5;
    const int lane   = tid & 31;
    const int laneHi = lane >> 4;
    const int lm     = lane & 15;
    const int wrow   = wave * 16;
    const int r0     = laneHi ? 8 : 0;

    f8_t acc[4];
    #pragma unroll
    for (int ct = 0; ct < 4; ++ct) acc[ct] = (f8_t)0.0f;

    const int arow = wrow + lm;   // A-matrix row for this lane

    // Loop over the two 32-wide halves of each 64-wide input section.
    // One q/f LDS read pair feeds A-fragments for all 4 sections (q, f,
    // q-f, q*f) -> 4x less A-side LDS traffic; unroll 1 keeps VGPR
    // pressure below the 256-reg MSB threshold for occupancy.
    #pragma unroll 1
    for (int j = 0; j < 2; ++j) {
        const int kkb = j * 32;

        h8_t q0 = *(const h8_t*)(&Qs[arow][kkb + r0]);
        h8_t f0 = *(const h8_t*)(&Fs[arow][kkb + r0]);
        h8_t q1 = *(const h8_t*)(&Qs[arow][kkb + 16 + r0]);
        h8_t f1 = *(const h8_t*)(&Fs[arow][kkb + 16 + r0]);

        #pragma unroll
        for (int sec = 0; sec < 4; ++sec) {
            h8_t a0, a1;
            if      (sec == 0) { a0 = q0;        a1 = q1;        }
            else if (sec == 1) { a0 = f0;        a1 = f1;        }
            else if (sec == 2) { a0 = q0 - f0;   a1 = q1 - f1;   }
            else               { a0 = q0 * f0;   a1 = q1 * f1;   }

            h16_t afrag;
            ((h8_t*)&afrag)[0] = a0;
            ((h8_t*)&afrag)[1] = a1;

            const int kb = sec * 64 + kkb;      // global K base for this step
            #pragma unroll
            for (int ct = 0; ct < 4; ++ct) {
                const int col = ct * 16 + lm;
                h16_t bfrag;
                ((h8_t*)&bfrag)[0] = *(const h8_t*)(&Wt[col][kb + r0]);
                ((h8_t*)&bfrag)[1] = *(const h8_t*)(&Wt[col][kb + 16 + r0]);
                acc[ct] = __builtin_amdgcn_wmma_f32_16x16x32_f16(
                    false, afrag, false, bfrag, (short)0, acc[ct], false, false);
            }
        }
    }

    __syncthreads();   // all Fs/Qs fragment reads complete before reuse

    // epilogue: bias, BN stats in f32, f16 tile into LDS (reuse Fs)
    #pragma unroll
    for (int ct = 0; ct < 4; ++ct) {
        const int col = ct * 16 + lm;
        const float bb = bias1[col];
        #pragma unroll
        for (int v = 0; v < 8; ++v) {
            const int row = wrow + v + laneHi * 8;
            const float h = acc[ct][v] + bb;
            Fs[row][col] = (_Float16)h;
            lds_fadd(&ssum[col], h);
            lds_fadd(&ssq[col], h * h);
        }
    }
    __syncthreads();

    // coalesced 16B stores of the f16 tile
    for (int i = tid; i < ROWS_PER_BLOCK * 8; i += 256) {
        int r  = i >> 3;
        int c8 = (i & 7) * 8;
        *(h8_t*)(h1 + (size_t)(n0 + r) * H1 + c8) = *(const h8_t*)(&Fs[r][c8]);
    }
    if (tid < H1) {
        glb_fadd(&stats1[tid], ssum[tid]);
        glb_fadd(&stats1[H1 + tid], ssq[tid]);
    }
}

// -------------------------------------------------------------------------
// finalize BN+Dice stats into per-feature affine coeffs:
//   y = A*h + B   (BatchNorm output)
//   gate_in = C*h + D  (Dice's BN applied to y, derived analytically:
//                       mean(y)=beta, var(y)=A^2*v)
// -------------------------------------------------------------------------
__global__ void finalize_kernel(
    const float* __restrict__ stats,
    const float* __restrict__ g, const float* __restrict__ be,
    const float* __restrict__ dg, const float* __restrict__ db,
    float* __restrict__ coeff, int H, float invN)
{
    int j = threadIdx.x;
    if (j >= H) return;
    float m = stats[j] * invN;
    float v = stats[H + j] * invN - m * m;
    float rstd = rsqrtf(v + 1e-5f);
    float A = g[j] * rstd;
    float B = be[j] - A * m;
    float varY = A * A * v;                 // var of BN output
    float rstdD = rsqrtf(varY + 1e-8f);
    float C = dg[j] * A * rstdD;
    float D = db[j] - C * m;
    coeff[j] = A; coeff[H + j] = B; coeff[2 * H + j] = C; coeff[3 * H + j] = D;
}

// -------------------------------------------------------------------------
// GEMM2: x = dice1(h1) (per-feature affine + sigmoid), h2 = x @ W2 + b2
// h1/h2 are f16 in global memory; activation math in f32.
// -------------------------------------------------------------------------
__global__ __launch_bounds__(256) void gemm2_kernel(
    const _Float16* __restrict__ h1, const float* __restrict__ W2,
    const float* __restrict__ bias2, const float* __restrict__ coeff1,
    const float* __restrict__ alpha1,
    _Float16* __restrict__ h2, float* __restrict__ stats2)
{
    __shared__ _Float16 Xs[ROWS_PER_BLOCK][72];   // activated tile, reused for output
    __shared__ _Float16 Wt[H2][72];
    __shared__ float cA[H1], cB[H1], cC[H1], cD[H1], cal[H1];
    __shared__ float ssum[H2];
    __shared__ float ssq[H2];

    const int tid = threadIdx.x;
    const int n0  = blockIdx.x * ROWS_PER_BLOCK;

    __builtin_prefetch(h1 + (size_t)n0 * H1, 0, 0);

    if (tid < H1) {
        cA[tid] = coeff1[tid];
        cB[tid] = coeff1[H1 + tid];
        cC[tid] = coeff1[2 * H1 + tid];
        cD[tid] = coeff1[3 * H1 + tid];
        cal[tid] = alpha1[tid];
    }
    if (tid < H2) { ssum[tid] = 0.0f; ssq[tid] = 0.0f; }
    __syncthreads();

    // load f16 h1 tile (16B vectors), apply BN+Dice in f32, back to f16
    for (int i = tid; i < ROWS_PER_BLOCK * 8; i += 256) {
        int r  = i >> 3;
        int c8 = (i & 7) * 8;
        h8_t hv = *(const h8_t*)(h1 + (size_t)(n0 + r) * H1 + c8);
        h8_t xv;
        #pragma unroll
        for (int u = 0; u < 8; ++u) {
            int cc = c8 + u;
            float h = (float)hv[u];
            float y = cA[cc] * h + cB[cc];
            float z = cC[cc] * h + cD[cc];
            float gte = 1.0f / (1.0f + __expf(-z));
            float al = cal[cc];
            xv[u] = (_Float16)(y * (al + (1.0f - al) * gte));
        }
        *(h8_t*)(&Xs[r][c8]) = xv;
    }
    for (int i = tid; i < H1 * H2; i += 256) {
        int k = i >> 5;
        int c = i & 31;
        Wt[c][k] = (_Float16)W2[i];
    }
    __syncthreads();

    const int wave   = tid >> 5;
    const int lane   = tid & 31;
    const int laneHi = lane >> 4;
    const int lm     = lane & 15;
    const int wrow   = wave * 16;
    const int r0     = laneHi ? 8 : 0;
    const int arow   = wrow + lm;

    f8_t acc[2];
    acc[0] = (f8_t)0.0f;
    acc[1] = (f8_t)0.0f;

    #pragma unroll
    for (int t = 0; t < 2; ++t) {           // K=64 -> 2 steps of 32
        const int kb = t * 32;
        h16_t afrag;
        ((h8_t*)&afrag)[0] = *(const h8_t*)(&Xs[arow][kb + r0]);
        ((h8_t*)&afrag)[1] = *(const h8_t*)(&Xs[arow][kb + 16 + r0]);
        #pragma unroll
        for (int ct = 0; ct < 2; ++ct) {
            const int col = ct * 16 + lm;
            h16_t bfrag;
            ((h8_t*)&bfrag)[0] = *(const h8_t*)(&Wt[col][kb + r0]);
            ((h8_t*)&bfrag)[1] = *(const h8_t*)(&Wt[col][kb + 16 + r0]);
            acc[ct] = __builtin_amdgcn_wmma_f32_16x16x32_f16(
                false, afrag, false, bfrag, (short)0, acc[ct], false, false);
        }
    }

    __syncthreads();   // all Xs fragment reads complete before reuse

    #pragma unroll
    for (int ct = 0; ct < 2; ++ct) {
        const int col = ct * 16 + lm;
        const float bb = bias2[col];
        #pragma unroll
        for (int v = 0; v < 8; ++v) {
            const int row = wrow + v + laneHi * 8;
            const float h = acc[ct][v] + bb;
            Xs[row][col] = (_Float16)h;
            lds_fadd(&ssum[col], h);
            lds_fadd(&ssq[col], h * h);
        }
    }
    __syncthreads();

    for (int i = tid; i < ROWS_PER_BLOCK * 4; i += 256) {
        int r  = i >> 2;
        int c8 = (i & 3) * 8;
        *(h8_t*)(h2 + (size_t)(n0 + r) * H2 + c8) = *(const h8_t*)(&Xs[r][c8]);
    }
    if (tid < H2) {
        glb_fadd(&stats2[tid], ssum[tid]);
        glb_fadd(&stats2[H2 + tid], ssq[tid]);
    }
}

// -------------------------------------------------------------------------
// per-batch: dice2 + final fc -> scores, mask, softmax over S, weighted sum
// -------------------------------------------------------------------------
__global__ __launch_bounds__(256) void attn_kernel(
    const _Float16* __restrict__ h2, const float* __restrict__ facts,
    const int* __restrict__ mask, const float* __restrict__ coeff2,
    const float* __restrict__ alpha2, const float* __restrict__ Wf,
    const float* __restrict__ bf, float* __restrict__ out)
{
    __shared__ float sc[256];
    __shared__ float wgt[SEQ];
    __shared__ float red[256];
    __shared__ float cA[H2], cB[H2], cC[H2], cD[H2], cal[H2], cW[H2];

    const int b   = blockIdx.x;
    const int tid = threadIdx.x;

    __builtin_prefetch(facts + (size_t)b * SEQ * EMB, 0, 0);

    if (tid < H2) {
        cA[tid] = coeff2[tid];
        cB[tid] = coeff2[H2 + tid];
        cC[tid] = coeff2[2 * H2 + tid];
        cD[tid] = coeff2[3 * H2 + tid];
        cal[tid] = alpha2[tid];
        cW[tid] = Wf[tid];
    }
    __syncthreads();

    float score = -3.0e38f;
    if (tid < SEQ) {
        const _Float16* hp = h2 + ((size_t)b * SEQ + tid) * H2;
        h8_t hv[4];
        #pragma unroll
        for (int k = 0; k < 4; ++k) hv[k] = *(const h8_t*)(hp + k * 8);
        float s = bf[0];
        #pragma unroll
        for (int j = 0; j < H2; ++j) {
            float h = (float)hv[j >> 3][j & 7];
            float y = cA[j] * h + cB[j];
            float z = cC[j] * h + cD[j];
            float gte = 1.0f / (1.0f + __expf(-z));
            float al = cal[j];
            s += y * (al + (1.0f - al) * gte) * cW[j];
        }
        if (mask[b * SEQ + tid] <= 0) s = -1e9f;
        score = s;
    }
    sc[tid] = score;
    red[tid] = score;
    __syncthreads();

    // max reduction
    #pragma unroll
    for (int off = 128; off > 0; off >>= 1) {
        if (tid < off) red[tid] = fmaxf(red[tid], red[tid + off]);
        __syncthreads();
    }
    const float mx = red[0];
    __syncthreads();

    const float ev = (tid < SEQ) ? __expf(sc[tid] - mx) : 0.0f;
    red[tid] = ev;
    __syncthreads();
    #pragma unroll
    for (int off = 128; off > 0; off >>= 1) {
        if (tid < off) red[tid] += red[tid + off];
        __syncthreads();
    }
    const float inv = 1.0f / red[0];
    __syncthreads();

    if (tid < SEQ) wgt[tid] = ev * inv;
    __syncthreads();

    // weighted sum over facts: 64 cols x 4 seq-groups of 50
    const int e   = tid & 63;
    const int grp = tid >> 6;
    float acc = 0.0f;
    const int s0 = grp * 50;
    for (int s = s0; s < s0 + 50; ++s)
        acc += wgt[s] * facts[((size_t)b * SEQ + s) * EMB + e];
    red[tid] = acc;
    __syncthreads();
    if (tid < 64)
        out[(size_t)b * EMB + tid] = red[tid] + red[64 + tid] + red[128 + tid] + red[192 + tid];
}

// -------------------------------------------------------------------------
extern "C" void kernel_launch(void* const* d_in, const int* in_sizes, int n_in,
                              void* d_out, int out_size, void* d_ws, size_t ws_size,
                              hipStream_t stream) {
    const float* query = (const float*)d_in[0];
    const float* facts = (const float*)d_in[1];
    const int*   mask  = (const int*)d_in[2];
    const float* W1  = (const float*)d_in[3];
    const float* b1  = (const float*)d_in[4];
    const float* g1  = (const float*)d_in[5];
    const float* be1 = (const float*)d_in[6];
    const float* dg1 = (const float*)d_in[7];
    const float* db1 = (const float*)d_in[8];
    const float* a1  = (const float*)d_in[9];
    const float* W2  = (const float*)d_in[10];
    const float* b2  = (const float*)d_in[11];
    const float* g2  = (const float*)d_in[12];
    const float* be2 = (const float*)d_in[13];
    const float* dg2 = (const float*)d_in[14];
    const float* db2 = (const float*)d_in[15];
    const float* a2  = (const float*)d_in[16];
    const float* Wf  = (const float*)d_in[17];
    const float* bf  = (const float*)d_in[18];
    float* out = (float*)d_out;

    // workspace layout: f16 intermediates first, then f32 stats/coeffs
    _Float16* h1 = (_Float16*)d_ws;                     // NROWS*64 halves
    _Float16* h2 = h1 + (size_t)NROWS * H1;             // NROWS*32 halves
    float* stats1 = (float*)(h2 + (size_t)NROWS * H2);  // 128
    float* stats2 = stats1 + 2 * H1;                    // 64
    float* coeff1 = stats2 + 2 * H2;                    // 256
    float* coeff2 = coeff1 + 4 * H1;                    // 128

    hipMemsetAsync(stats1, 0, (2 * H1 + 2 * H2) * sizeof(float), stream);

    const int nblk = NROWS / ROWS_PER_BLOCK;            // 6400
    gemm1_kernel<<<nblk, 256, 0, stream>>>(query, facts, W1, b1, h1, stats1);
    finalize_kernel<<<1, 64, 0, stream>>>(stats1, g1, be1, dg1, db1, coeff1, H1, 1.0f / (float)NROWS);
    gemm2_kernel<<<nblk, 256, 0, stream>>>(h1, W2, b2, coeff1, a1, h2, stats2);
    finalize_kernel<<<1, 32, 0, stream>>>(stats2, g2, be2, dg2, db2, coeff2, H2, 1.0f / (float)NROWS);
    attn_kernel<<<BATCH, 256, 0, stream>>>(h2, facts, mask, coeff2, a2, Wf, bf, out);
}
